// Transform_16509854286692
// MI455X (gfx1250) — matrix operand order repaired and checked by reading, not compiled, not run on hardware
//
#include <hip/hip_runtime.h>

// ---------------------------------------------------------------------------
// Transformer block on MI455X (gfx1250): bf16 WMMA everywhere.
// x:(2,8,1024,512) fp32. QKV proj -> 8-head attention over N=1024 ->
// residual+LN -> FFN(relu) -> LN.  ~77 GFLOP, all GEMMs via
// v_wmma_f32_16x16x32_bf16 with fp32 accumulation.
// GEMM waves use 32x64 tiles: 8 WMMAs per 6 fragment loads.
// ---------------------------------------------------------------------------

typedef __bf16 bf16;
typedef __attribute__((ext_vector_type(16))) __bf16 v16bf;
typedef __attribute__((ext_vector_type(8)))  __bf16 v8bf;
typedef __attribute__((ext_vector_type(8)))  float  v8f;

#define WMMA_BF16(a, b, c) \
  __builtin_amdgcn_wmma_f32_16x16x32_bf16(false, (a), false, (b), (short)0, (c), false, false)

static __device__ __forceinline__ v8f vzero8() {
  v8f r;
#pragma unroll
  for (int i = 0; i < 8; ++i) r[i] = 0.0f;
  return r;
}

// A-fragment (16x32, bf16) from row-major matrix with row stride ld (elems).
// Lane layout (ISA 7.12.2): row m = lane%16; k = (e&7) + 8*(2*(e>>3) + lane/16).
static __device__ __forceinline__ v16bf load_a_frag(const bf16* A, int ld, int m0, int k0) {
  const int lane = threadIdx.x & 31;
  const int hl   = lane >> 4;
  const bf16* p  = A + (size_t)(m0 + (lane & 15)) * ld + k0 + 8 * hl;
  v8bf lo = *(const v8bf*)p;          // k = k0 + 8*hl  + [0..7]
  v8bf hi = *(const v8bf*)(p + 16);   // k = k0 + 16 + 8*hl + [0..7]
  v16bf r;
#pragma unroll
  for (int i = 0; i < 8; ++i) { r[i] = lo[i]; r[i + 8] = hi[i]; }
  return r;
}

// B-fragment (32x16, bf16) from a "column-contiguous" matrix Bt where
// Bt[n][k] is row-major with stride ld (i.e. column n of B contiguous in k).
// Lane layout: col n = lane%16; k = (e&7) + 8*(2*(e>>3) + lane/16)  -> two 16B loads.
static __device__ __forceinline__ v16bf load_bt_frag(const bf16* Bt, int ld, int k0, int n0) {
  const int lane = threadIdx.x & 31;
  const bf16* p  = Bt + (size_t)(n0 + (lane & 15)) * ld + k0 + 8 * (lane >> 4);
  v8bf lo = *(const v8bf*)p;
  v8bf hi = *(const v8bf*)(p + 16);
  v16bf r;
#pragma unroll
  for (int i = 0; i < 8; ++i) { r[i] = lo[i]; r[i + 8] = hi[i]; }
  return r;
}

// ---------------------------------------------------------------------------
// Elementwise converts / transposes
// ---------------------------------------------------------------------------
__global__ __launch_bounds__(256) void cvt_f32_bf16_kernel(const float* __restrict__ in,
                                                           bf16* __restrict__ out) {
  const int i = blockIdx.x * 256 + threadIdx.x;
  out[i] = (bf16)in[i];
}

// W (K x N, fp32, row-major) -> Wt (N x K, bf16)
__global__ __launch_bounds__(256) void wtrans_kernel(const float* __restrict__ W,
                                                     bf16* __restrict__ Wt, int K, int N) {
  const int idx = blockIdx.x * 256 + threadIdx.x;
  const int k = idx / N, n = idx % N;
  Wt[(size_t)n * K + k] = (bf16)W[idx];
}

// V[(bt*1024+j)*512 + head*64 + d] -> Vt[((bt*8+head)*64 + d)*1024 + j]
__global__ __launch_bounds__(256) void vtrans_kernel(const bf16* __restrict__ V,
                                                     bf16* __restrict__ Vt) {
  const int idx  = blockIdx.x * 256 + threadIdx.x;   // over 16384*512
  const int row  = idx >> 9;
  const int col  = idx & 511;
  const int bt   = row >> 10;
  const int j    = row & 1023;
  const int head = col >> 6;
  const int d    = col & 63;
  Vt[(size_t)((bt * 8 + head) * 64 + d) * 1024 + j] = V[idx];
}

// ---------------------------------------------------------------------------
// GEMM: out = A(MxK,bf16) * W(KxN) + bias, W given transposed as Wt(NxK,bf16).
// 128 threads = 4 waves; wave w computes rows [blockIdx.x*128 + 32w, +32),
// cols [blockIdx.y*64, +64): 2 A-frags x 4 B-frags = 8 WMMAs per k-chunk,
// each B-frag feeding two WMMAs.
// ---------------------------------------------------------------------------
__global__ __launch_bounds__(128) void gemm_bf16_kernel(
    const bf16* __restrict__ A, const bf16* __restrict__ Wt,
    const float* __restrict__ bias, float* __restrict__ outF,
    bf16* __restrict__ outB, int M, int N, int K, int relu) {
  const int lane = threadIdx.x & 31;
  const int wave = threadIdx.x >> 5;
  const int hl   = lane >> 4;
  const int m0   = blockIdx.x * 128 + wave * 32;
  const int n0   = blockIdx.y * 64;

  v8f acc[2][4];
#pragma unroll
  for (int r = 0; r < 2; ++r)
#pragma unroll
    for (int c = 0; c < 4; ++c) acc[r][c] = vzero8();

  for (int k0 = 0; k0 < K; k0 += 32) {
    const v16bf aF0 = load_a_frag(A, K, m0,      k0);
    const v16bf aF1 = load_a_frag(A, K, m0 + 16, k0);
#pragma unroll
    for (int c = 0; c < 4; ++c) {
      const v16bf bF = load_bt_frag(Wt, K, k0, n0 + 16 * c);
      acc[0][c] = WMMA_BF16(aF0, bF, acc[0][c]);
      acc[1][c] = WMMA_BF16(aF1, bF, acc[1][c]);
    }
  }

#pragma unroll
  for (int r = 0; r < 2; ++r) {
#pragma unroll
    for (int c = 0; c < 4; ++c) {
#pragma unroll
      for (int e = 0; e < 8; ++e) {
        const int m = m0 + 16 * r + e + 8 * hl;
        const int n = n0 + 16 * c + (lane & 15);
        float v = acc[r][c][e] + bias[n];
        if (relu) v = fmaxf(v, 0.0f);
        if (outF) outF[(size_t)m * N + n] = v;
        if (outB) outB[(size_t)m * N + n] = (bf16)v;
      }
    }
  }
}

// ---------------------------------------------------------------------------
// Flash attention per (bt, head): softmax(Q K^T / 8) V over N=1024 keys.
// 256 threads = 8 waves; wave owns 16 query rows; keys in chunks of 32 with
// online softmax.  P transits LDS for C-layout -> A-layout transpose.
// ---------------------------------------------------------------------------
__global__ __launch_bounds__(256) void attn_kernel(const bf16* __restrict__ Qb,
                                                   const bf16* __restrict__ Kb,
                                                   const bf16* __restrict__ Vt,
                                                   float* __restrict__ O) {
  __shared__ __align__(32) bf16 plds[8 * 16 * 32];

  const int lane = threadIdx.x & 31;
  const int wave = threadIdx.x >> 5;
  const int hl   = lane >> 4;
  const int bt   = blockIdx.y >> 3;
  const int head = blockIdx.y & 7;
  const int i0   = (blockIdx.x * 8 + wave) * 16;

  const bf16* Qbase = Qb + (size_t)bt * 1024 * 512 + head * 64;
  const bf16* Kbase = Kb + (size_t)bt * 1024 * 512 + head * 64;
  const bf16* Vbase = Vt + (size_t)((bt * 8 + head) * 64) * 1024;
  bf16* pl = plds + wave * (16 * 32);

  const v16bf q0 = load_a_frag(Qbase, 512, i0, 0);
  const v16bf q1 = load_a_frag(Qbase, 512, i0, 32);

  v8f acc[4];
#pragma unroll
  for (int c = 0; c < 4; ++c) acc[c] = vzero8();
  float mi[8], li[8];
#pragma unroll
  for (int e = 0; e < 8; ++e) { mi[e] = -1e30f; li[e] = 0.0f; }

  for (int j0 = 0; j0 < 1024; j0 += 32) {
    // S tiles for keys [j0,j0+16) and [j0+16,j0+32): accumulate over d=64.
    v8f s0 = vzero8(), s1 = vzero8();
    s0 = WMMA_BF16(q0, load_bt_frag(Kbase, 512, 0,  j0),      s0);
    s0 = WMMA_BF16(q1, load_bt_frag(Kbase, 512, 32, j0),      s0);
    s1 = WMMA_BF16(q0, load_bt_frag(Kbase, 512, 0,  j0 + 16), s1);
    s1 = WMMA_BF16(q1, load_bt_frag(Kbase, 512, 32, j0 + 16), s1);

    float corr[8];
#pragma unroll
    for (int e = 0; e < 8; ++e) {
      const float a  = s0[e] * 0.125f;   // 1/sqrt(64)
      const float b2 = s1[e] * 0.125f;
      // row max across the 16 lanes of this half (rows m = e + 8*hl)
      float rm = fmaxf(a, b2);
#pragma unroll
      for (int off = 1; off < 16; off <<= 1) rm = fmaxf(rm, __shfl_xor(rm, off, 32));
      const float nm = fmaxf(mi[e], rm);
      corr[e] = __expf(mi[e] - nm);
      mi[e]   = nm;
      const float p0 = __expf(a - nm);
      const float p1 = __expf(b2 - nm);
      float rs = p0 + p1;
#pragma unroll
      for (int off = 1; off < 16; off <<= 1) rs += __shfl_xor(rs, off, 32);
      li[e] = li[e] * corr[e] + rs;
      // stash P (16 queries x 32 keys) in LDS, C-layout write
      const int m = e + 8 * hl;
      pl[m * 32 + (lane & 15)]      = (bf16)p0;
      pl[m * 32 + 16 + (lane & 15)] = (bf16)p1;
    }
    asm volatile("s_wait_dscnt 0" ::: "memory");   // LDS RAW within wave
    const v16bf pf = load_a_frag(pl, 32, 0, 0);    // A-layout read-back

#pragma unroll
    for (int c = 0; c < 4; ++c) {
#pragma unroll
      for (int e = 0; e < 8; ++e) acc[c][e] *= corr[e];
      acc[c] = WMMA_BF16(pf, load_bt_frag(Vbase, 1024, j0, 16 * c), acc[c]);
    }
  }

#pragma unroll
  for (int c = 0; c < 4; ++c) {
#pragma unroll
    for (int e = 0; e < 8; ++e) {
      const int m = i0 + e + 8 * hl;
      const int n = head * 64 + 16 * c + (lane & 15);
      O[((size_t)bt * 1024 + m) * 512 + n] = acc[c][e] / li[e];
    }
  }
}

// ---------------------------------------------------------------------------
// LayerNorm over F=512 (biased variance, eps=1e-5), optional residual add,
// fp32 and/or bf16 output.  1 wave per row, 8 rows per block.
// ---------------------------------------------------------------------------
__global__ __launch_bounds__(256) void ln_kernel(const float* __restrict__ X,
                                                 const float* __restrict__ resid,
                                                 const float* __restrict__ g,
                                                 const float* __restrict__ b,
                                                 float* __restrict__ outF,
                                                 bf16* __restrict__ outB) {
  const int lane = threadIdx.x & 31;
  const int wave = threadIdx.x >> 5;
  const int row  = blockIdx.x * 8 + wave;
  const float* xr = X + (size_t)row * 512;
  const float* rr = resid ? resid + (size_t)row * 512 : nullptr;

  float v[16], s = 0.0f, sq = 0.0f;
#pragma unroll
  for (int j = 0; j < 16; ++j) {
    const int idx = j * 32 + lane;
    float t = xr[idx];
    if (rr) t += rr[idx];
    v[j] = t; s += t; sq += t * t;
  }
#pragma unroll
  for (int off = 1; off < 32; off <<= 1) {
    s  += __shfl_xor(s,  off, 32);
    sq += __shfl_xor(sq, off, 32);
  }
  const float mean = s * (1.0f / 512.0f);
  const float var  = sq * (1.0f / 512.0f) - mean * mean;
  const float rstd = rsqrtf(var + 1e-5f);
#pragma unroll
  for (int j = 0; j < 16; ++j) {
    const int idx = j * 32 + lane;
    const float y = (v[j] - mean) * rstd * g[idx] + b[idx];
    if (outF) outF[(size_t)row * 512 + idx] = y;
    if (outB) outB[(size_t)row * 512 + idx] = (bf16)y;
  }
}

// ---------------------------------------------------------------------------
// Host-side orchestration
// ---------------------------------------------------------------------------
extern "C" void kernel_launch(void* const* d_in, const int* in_sizes, int n_in,
                              void* d_out, int out_size, void* d_ws, size_t ws_size,
                              hipStream_t stream) {
  (void)in_sizes; (void)n_in; (void)out_size; (void)ws_size;
  const float* x    = (const float*)d_in[0];
  const float* Wq   = (const float*)d_in[1];
  const float* bq   = (const float*)d_in[2];
  const float* Wk   = (const float*)d_in[3];
  const float* bk   = (const float*)d_in[4];
  const float* Wv   = (const float*)d_in[5];
  const float* bv   = (const float*)d_in[6];
  const float* ln_g = (const float*)d_in[7];
  const float* ln_b = (const float*)d_in[8];
  const float* W1   = (const float*)d_in[9];
  const float* b1   = (const float*)d_in[10];
  const float* W2   = (const float*)d_in[11];
  const float* b2   = (const float*)d_in[12];
  const float* lg   = (const float*)d_in[13];
  const float* lb   = (const float*)d_in[14];
  float* out = (float*)d_out;

  const int ROWS = 16384;            // B*T*N
  const int F    = 512;
  const size_t MAT_B = (size_t)ROWS * F * sizeof(bf16);   // 16 MB
  const size_t W_B   = (size_t)F * F * sizeof(bf16);      // 512 KB

  char* ws = (char*)d_ws;
  bf16* xb   = (bf16*)(ws);                         // reused later as ff1_b
  bf16* Wqt  = (bf16*)(ws + MAT_B);
  bf16* Wkt  = (bf16*)(ws + MAT_B + 1 * W_B);
  bf16* Wvt  = (bf16*)(ws + MAT_B + 2 * W_B);
  bf16* W1t  = (bf16*)(ws + MAT_B + 3 * W_B);
  bf16* W2t  = (bf16*)(ws + MAT_B + 4 * W_B);
  char* p    = ws + MAT_B + 5 * W_B;
  bf16* q_b  = (bf16*)(p);                          // reused later as val_b
  bf16* k_b  = (bf16*)(p + 1 * MAT_B);
  bf16* v_b  = (bf16*)(p + 2 * MAT_B);
  bf16* vt_b = (bf16*)(p + 3 * MAT_B);
  float* o_f = (float*)(p + 4 * MAT_B);             // reused later as ff2_f
  bf16* val_b = q_b;
  bf16* ff1_b = xb;
  float* ff2_f = o_f;

  const dim3 gemmGrid(ROWS / 128, F / 64);

  // 1) bf16 copies of activations and transposed weights
  cvt_f32_bf16_kernel<<<ROWS * F / 256, 256, 0, stream>>>(x, xb);
  wtrans_kernel<<<F * F / 256, 256, 0, stream>>>(Wq, Wqt, F, F);
  wtrans_kernel<<<F * F / 256, 256, 0, stream>>>(Wk, Wkt, F, F);
  wtrans_kernel<<<F * F / 256, 256, 0, stream>>>(Wv, Wvt, F, F);
  wtrans_kernel<<<F * F / 256, 256, 0, stream>>>(W1, W1t, F, F);
  wtrans_kernel<<<F * F / 256, 256, 0, stream>>>(W2, W2t, F, F);

  // 2) QKV projections (bf16 out for the attention WMMAs)
  gemm_bf16_kernel<<<gemmGrid, 128, 0, stream>>>(xb, Wqt, bq, nullptr, q_b, ROWS, F, F, 0);
  gemm_bf16_kernel<<<gemmGrid, 128, 0, stream>>>(xb, Wkt, bk, nullptr, k_b, ROWS, F, F, 0);
  gemm_bf16_kernel<<<gemmGrid, 128, 0, stream>>>(xb, Wvt, bv, nullptr, v_b, ROWS, F, F, 0);

  // 3) V transpose so the P*V B-fragments are contiguous loads
  vtrans_kernel<<<ROWS * F / 256, 256, 0, stream>>>(v_b, vt_b);

  // 4) attention: grid = (1024/(16*8), B*T*heads)
  attn_kernel<<<dim3(8, 128), 256, 0, stream>>>(q_b, k_b, vt_b, o_f);

  // 5) residual + LN1 -> bf16 activations for FFN
  ln_kernel<<<ROWS / 8, 256, 0, stream>>>(o_f, x, ln_g, ln_b, nullptr, val_b);

  // 6) FFN: relu(val@W1+b1) @ W2 + b2
  gemm_bf16_kernel<<<gemmGrid, 128, 0, stream>>>(val_b, W1t, b1, nullptr, ff1_b, ROWS, F, F, 1);
  gemm_bf16_kernel<<<gemmGrid, 128, 0, stream>>>(ff1_b, W2t, b2, ff2_f, nullptr, ROWS, F, F, 0);

  // 7) final LN -> fp32 output
  ln_kernel<<<ROWS / 8, 256, 0, stream>>>(ff2_f, nullptr, lg, lb, out, nullptr);
}
